// Model_58102317580662
// MI455X (gfx1250) — matrix operand order repaired and checked by reading, not compiled, not run on hardware
//
#include <hip/hip_runtime.h>
#include <hip/hip_bf16.h>
#include <math.h>

typedef __attribute__((ext_vector_type(16))) _Float16 v16h;
typedef __attribute__((ext_vector_type(8)))  _Float16 v8h;
typedef __attribute__((ext_vector_type(8)))  float    v8f;

#define NTOT 1024   // nodes per batch (N1+N2)
#define BS2  2
#define D    64

// Branch-free tanh: prefer gfx1250 hardware v_tanh_f32, else 1 exp + rcp.
__device__ __forceinline__ float fast_tanh(float x) {
#if __has_builtin(__builtin_amdgcn_tanhf)
    return __builtin_amdgcn_tanhf(x);
#else
    float ax = __builtin_fabsf(x);
    float e  = __expf(-2.0f * ax);
    float r  = (1.0f - e) * __builtin_amdgcn_rcpf(1.0f + e);
    return __builtin_copysignf(r, x);
#endif
}

// ---------------------------------------------------------------------------
// Kernel 1: x = concat(x1@W1+b1, x2@W2+b2); also emit f16 copy for WMMA feed
// ---------------------------------------------------------------------------
__global__ void k_proj(const float* __restrict__ x1, const float* __restrict__ x2,
                       const float* __restrict__ W1, const float* __restrict__ b1,
                       const float* __restrict__ W2, const float* __restrict__ b2,
                       float* __restrict__ X, _Float16* __restrict__ Xh) {
    int row = blockIdx.x;              // 0..2047  = b*1024 + n
    int b = row >> 10, n = row & 1023;
    const float *xin, *W, *bias;
    if (n < 512) { xin = x1 + (b * 512 + n) * D;         W = W1; bias = b1; }
    else         { xin = x2 + (b * 512 + (n - 512)) * D; W = W2; bias = b2; }
    __shared__ float xr[D];
    int d = threadIdx.x;
    xr[d] = xin[d];
    __syncthreads();
    float acc = bias[d];
#pragma unroll
    for (int k = 0; k < D; ++k) acc += xr[k] * W[k * D + d];
    X[row * D + d]  = acc;
    Xh[row * D + d] = (_Float16)acc;
}

// ---------------------------------------------------------------------------
// Kernel 2: fused attention row.  One workgroup (256 thr = 8 waves) per (b,i).
//   B' = x_i (*) Wa  in LDS (f16, transposed);  stream 1024 x_j rows via WMMA;
//   tanh + block score weights -> scores[j];  softmax over j;  agg_i = sum.
// ---------------------------------------------------------------------------
__global__ void __launch_bounds__(256)
k_att(const float* __restrict__ X, const _Float16* __restrict__ Xh,
      const float* __restrict__ Wa, const float* __restrict__ ba,
      const float* __restrict__ w11, const float* __restrict__ w22,
      const float* __restrict__ w12, float* __restrict__ AGG) {
    const int row  = blockIdx.x;        // b*1024 + i
    const int b    = row >> 10;
    const int i    = row & 1023;
    const int tid  = threadIdx.x;
    const int lane = tid & 31;
    const int wave = tid >> 5;
    const int half = lane >> 4;
    const int l15  = lane & 15;

    __shared__ _Float16 BpT[D * D];     // 8 KB: B'[k][n] stored as BpT[n*64+k]
    __shared__ float    sc[NTOT];       // 4 KB: scores / probs
    __shared__ float    xi[D];
    __shared__ float    red[256];

    const float*    Xb  = X  + b * NTOT * D;
    const _Float16* Xhb = Xh + b * NTOT * D;

    if (tid < D) xi[tid] = Xb[i * D + tid];
    __syncthreads();
    // coalesced Wa read; transposed f16 store (one-time per workgroup)
#pragma unroll
    for (int t = 0; t < 16; ++t) {
        int idx = t * 256 + tid;
        int k = idx >> 6, n = idx & 63;
        BpT[n * D + k] = (_Float16)(xi[k] * Wa[idx]);
    }
    __syncthreads();

    // --- preload B fragments (contiguous 16-half runs -> 2 x b128 each) ---
    v16h bf[2][4];
#pragma unroll
    for (int kb = 0; kb < 2; ++kb)
#pragma unroll
        for (int c = 0; c < 4; ++c) {
            const v8h* p = (const v8h*)&BpT[(c * 16 + l15) * D + kb * 32 + half * 16];
            bf[kb][c] = __builtin_shufflevector(p[0], p[1],
                            0, 1, 2, 3, 4, 5, 6, 7, 8, 9, 10, 11, 12, 13, 14, 15);
        }

    // score-weight vectors by board block (i-half fixed per workgroup)
    const float* wlo = (i < 512) ? w11 : w12;   // j < 512
    const float* whi = (i < 512) ? w12 : w22;   // j >= 512
    float wl[4], wh[4], bav[4];
#pragma unroll
    for (int c = 0; c < 4; ++c) {
        wl[c]  = wlo[c * 16 + l15];
        wh[c]  = whi[c * 16 + l15];
        bav[c] = ba[c * 16 + l15];
    }

    // --- stream all j rows: each wave owns 128 rows (8 tiles of 16) -------
#pragma unroll
    for (int it = 0; it < 8; ++it) {
        int j0 = wave * 128 + it * 16;
        const v8h* xh = (const v8h*)(Xhb + (j0 + l15) * D);  // 8-half chunks
        v16h a0 = __builtin_shufflevector(xh[2 * half],     xh[2 * half + 1],
                      0, 1, 2, 3, 4, 5, 6, 7, 8, 9, 10, 11, 12, 13, 14, 15);
        v16h a1 = __builtin_shufflevector(xh[4 + 2 * half], xh[5 + 2 * half],
                      0, 1, 2, 3, 4, 5, 6, 7, 8, 9, 10, 11, 12, 13, 14, 15);
        float p[8];
#pragma unroll
        for (int r = 0; r < 8; ++r) p[r] = 0.f;
        const bool lohalf = (j0 < 512);
#pragma unroll
        for (int c = 0; c < 4; ++c) {
            v8f acc = {};
            acc = __builtin_amdgcn_wmma_f32_16x16x32_f16(
                      false, a0, false, bf[0][c], (short)0, acc, false, false);
            acc = __builtin_amdgcn_wmma_f32_16x16x32_f16(
                      false, a1, false, bf[1][c], (short)0, acc, false, false);
            float wv = lohalf ? wl[c] : wh[c];
#pragma unroll
            for (int r = 0; r < 8; ++r)
                p[r] += fast_tanh(acc[r] + bav[c]) * wv;
        }
        // reduce over the 16 lanes sharing a row half (xor stays in-half)
#pragma unroll
        for (int m = 1; m < 16; m <<= 1)
#pragma unroll
            for (int r = 0; r < 8; ++r)
                p[r] += __shfl_xor(p[r], m, 32);
        if (l15 == 0) {
#pragma unroll
            for (int r = 0; r < 8; ++r)
                sc[j0 + half * 8 + r] = p[r];
        }
    }
    __syncthreads();

    // --- softmax over j (1024) --------------------------------------------
    float lm = -3.4e38f;
    for (int j = tid; j < NTOT; j += 256) lm = fmaxf(lm, sc[j]);
    red[tid] = lm; __syncthreads();
    for (int s = 128; s > 0; s >>= 1) {
        if (tid < s) red[tid] = fmaxf(red[tid], red[tid + s]);
        __syncthreads();
    }
    float mx = red[0];
    __syncthreads();
    float ls = 0.f;
    for (int j = tid; j < NTOT; j += 256) {
        float e = __expf(sc[j] - mx);
        sc[j] = e;
        ls += e;
    }
    red[tid] = ls; __syncthreads();
    for (int s = 128; s > 0; s >>= 1) {
        if (tid < s) red[tid] += red[tid + s];
        __syncthreads();
    }
    float S = red[0];
    __syncthreads();

    // --- agg_i = sum_j att_ij * x_j ---------------------------------------
    int d = tid & 63, g = tid >> 6;
    float acc = 0.f;
    for (int j = g * 256; j < g * 256 + 256; ++j)
        acc += sc[j] * Xb[j * D + d];
    red[tid] = acc; __syncthreads();
    if (tid < D) {
        float tot = red[tid] + red[tid + 64] + red[tid + 128] + red[tid + 192];
        AGG[row * D + tid] = tot / S;
    }
}

// ---------------------------------------------------------------------------
// Kernel 3: xo = agg@Wpa + bpa + x@Wwoa + bwoa
// ---------------------------------------------------------------------------
__global__ void k_out(const float* __restrict__ X, const float* __restrict__ AGG,
                      const float* __restrict__ Wpa, const float* __restrict__ bpa,
                      const float* __restrict__ Wwoa, const float* __restrict__ bwoa,
                      float* __restrict__ XO) {
    int row = blockIdx.x, d = threadIdx.x;
    __shared__ float ar[D], xr[D];
    ar[d] = AGG[row * D + d];
    xr[d] = X[row * D + d];
    __syncthreads();
    float acc = bpa[d] + bwoa[d];
#pragma unroll
    for (int k = 0; k < D; ++k)
        acc += ar[k] * Wpa[k * D + d] + xr[k] * Wwoa[k * D + d];
    XO[row * D + d] = acc;
}

// ---------------------------------------------------------------------------
// Kernel 4: batchnorm stats over 2048 rows (single block, 1024 thr)
// ---------------------------------------------------------------------------
__global__ void __launch_bounds__(1024) k_stats(const float* __restrict__ XO,
                                                float* __restrict__ STATS) {
    int tid = threadIdx.x, d = tid & 63, g = tid >> 6;   // 16 row groups
    float s = 0.f, s2 = 0.f;
    for (int r = g; r < BS2 * NTOT; r += 16) {
        float v = XO[r * D + d];
        s += v; s2 += v * v;
    }
    __shared__ float rs[1024], rs2[1024];
    rs[tid] = s; rs2[tid] = s2; __syncthreads();
    for (int st = 512; st >= 64; st >>= 1) {
        if (tid < st) { rs[tid] += rs[tid + st]; rs2[tid] += rs2[tid + st]; }
        __syncthreads();
    }
    if (tid < D) {
        const float invn = 1.f / (float)(BS2 * NTOT);
        float mean = rs[tid] * invn;
        float var  = rs2[tid] * invn - mean * mean;
        STATS[tid]     = mean;
        STATS[D + tid] = rsqrtf(var + 1e-5f);
    }
}

// ---------------------------------------------------------------------------
// Kernel 5: normalize + SELU (in place) + pool score = sigmoid(xo@Wp + bp)
// ---------------------------------------------------------------------------
__global__ void k_norm(float* __restrict__ XO, const float* __restrict__ STATS,
                       const float* __restrict__ gamma, const float* __restrict__ beta,
                       const float* __restrict__ Wp, const float* __restrict__ bp,
                       float* __restrict__ SCORE) {
    int row = blockIdx.x, d = threadIdx.x;
    float v = XO[row * D + d];
    v = (v - STATS[d]) * STATS[D + d] * gamma[d] + beta[d];
    const float alpha = 1.6732632423543772f, lam = 1.0507009873554805f;
    v = lam * (v > 0.f ? v : alpha * (__expf(v) - 1.f));
    XO[row * D + d] = v;
    __shared__ float red[D];
    red[d] = v * Wp[d];
    __syncthreads();
    for (int s = 32; s > 0; s >>= 1) {
        if (d < s) red[d] += red[d + s];
        __syncthreads();
    }
    if (d == 0) SCORE[row] = 1.f / (1.f + __expf(-(red[0] + bp[0])));
}

// ---------------------------------------------------------------------------
// Kernel 6: stable top-k (k=256 of 512) via rank, gather xo*score
//   rank = #(s_j > s_i) + #(s_j == s_i && j < i)  -> matches lax.top_k order
// ---------------------------------------------------------------------------
__global__ void __launch_bounds__(512) k_pool(const float* __restrict__ XO,
                                              const float* __restrict__ SCORE,
                                              float* __restrict__ OUT) {
    int blk = blockIdx.x;              // 0..3 = b*2 + half
    int b = blk >> 1, h = blk & 1;
    int i = threadIdx.x;               // 0..511
    __shared__ float s[512];
    int node = b * NTOT + h * 512 + i;
    float si = SCORE[node];
    s[i] = si;
    __syncthreads();
    int rank = 0;
    for (int j = 0; j < 512; ++j) {
        float sj = s[j];
        rank += (sj > si) || (sj == si && j < i);
    }
    if (rank < 256) {
        float* o = OUT + ((b * 512) + h * 256 + rank) * D;
        const float* xr = XO + node * D;
#pragma unroll
        for (int d = 0; d < D; ++d) o[d] = xr[d] * si;
    }
}

// ---------------------------------------------------------------------------
extern "C" void kernel_launch(void* const* d_in, const int* in_sizes, int n_in,
                              void* d_out, int out_size, void* d_ws, size_t ws_size,
                              hipStream_t stream) {
    (void)in_sizes; (void)n_in; (void)out_size; (void)ws_size;
    const float* x1   = (const float*)d_in[0];
    const float* x2   = (const float*)d_in[1];
    const float* W1   = (const float*)d_in[2];
    const float* b1   = (const float*)d_in[3];
    const float* W2   = (const float*)d_in[4];
    const float* b2   = (const float*)d_in[5];
    const float* Wa   = (const float*)d_in[6];
    const float* ba   = (const float*)d_in[7];
    // d_in[8..9], [13], [18..21]: master branch — dead code in reference output
    const float* w11  = (const float*)d_in[10];
    const float* w22  = (const float*)d_in[11];
    const float* w12  = (const float*)d_in[12];
    const float* Wpa  = (const float*)d_in[14];
    const float* bpa  = (const float*)d_in[15];
    const float* Wwoa = (const float*)d_in[16];
    const float* bwoa = (const float*)d_in[17];
    const float* gamma= (const float*)d_in[22];
    const float* beta = (const float*)d_in[23];
    const float* Wp   = (const float*)d_in[24];
    const float* bp   = (const float*)d_in[25];
    float* OUT = (float*)d_out;

    float* ws    = (float*)d_ws;
    float* X     = ws;                     // 2*1024*64 f32
    float* AGG   = X    + BS2 * NTOT * D;  // 2*1024*64 f32
    float* XO    = AGG  + BS2 * NTOT * D;  // 2*1024*64 f32
    float* STATS = XO   + BS2 * NTOT * D;  // 128 f32
    float* SCORE = STATS + 2 * D;          // 2048 f32
    _Float16* Xh = (_Float16*)(SCORE + BS2 * NTOT);  // 2*1024*64 f16 (16B aligned)

    k_proj <<<BS2 * NTOT, D,    0, stream>>>(x1, x2, W1, b1, W2, b2, X, Xh);
    k_att  <<<BS2 * NTOT, 256,  0, stream>>>(X, Xh, Wa, ba, w11, w22, w12, AGG);
    k_out  <<<BS2 * NTOT, D,    0, stream>>>(X, AGG, Wpa, bpa, Wwoa, bwoa, XO);
    k_stats<<<1,          1024, 0, stream>>>(XO, STATS);
    k_norm <<<BS2 * NTOT, D,    0, stream>>>(XO, STATS, gamma, beta, Wp, bp, SCORE);
    k_pool <<<BS2 * 2,    512,  0, stream>>>(XO, SCORE, OUT);
}
